// PlasmidLMSparseMoE_20813411516960
// MI455X (gfx1250) — matrix operand order repaired and checked by reading, not compiled, not run on hardware
//
#include <hip/hip_runtime.h>
#include <math.h>

// ---------------- types ----------------
typedef __attribute__((ext_vector_type(16))) __bf16          bf16x16;
typedef __attribute__((ext_vector_type(8)))  float           f32x8;
typedef __attribute__((ext_vector_type(8)))  unsigned short  us8;
typedef __attribute__((ext_vector_type(4)))  unsigned short  us4;

// ---------------- problem constants ----------------
constexpr int NUM_EXPERTS = 8;
constexpr int TOPK        = 2;
constexpr int HIDDEN      = 1024;
constexpr int INTER       = 2048;
constexpr int NTOK        = 4096;            // BATCH*SEQ

// ---------------- GEMM tiling ----------------
constexpr int BM  = 256;                     // row tile (tokens)
constexpr int BN  = 128;                     // col tile
constexpr int BK  = 32;                      // == WMMA K for bf16
constexpr int LDA = 40;                      // LDS row stride (ushorts), keeps 16B-aligned frag starts
constexpr int LDB = 40;
constexpr int GEMM_THREADS = 512;            // 16 waves: 4 (M) x 4 (N)
constexpr int MAXROWS = 10240;               // 8192 pairs + 8*BM padding headroom

// ---------------- workspace layout (bytes) ----------------
constexpr size_t WS_CNT   = 0;                                    // int[8]  (also aux 'f' counts)
constexpr size_t WS_P     = 256;                                  // float[8]
constexpr size_t WS_OFF   = 512;                                  // int[9]
constexpr size_t WS_CUR   = 768;                                  // int[8]
constexpr size_t WS_TKI   = 1024;                                 // int[NTOK*2]
constexpr size_t WS_TKW   = WS_TKI  + (size_t)NTOK * 2 * 4;       // float[NTOK*2]
constexpr size_t WS_RTOK  = WS_TKW  + (size_t)NTOK * 2 * 4;       // int[MAXROWS]
constexpr size_t WS_RGATE = WS_RTOK + (size_t)MAXROWS * 4;        // float[MAXROWS]
constexpr size_t WS_HBUF  = WS_RGATE + (size_t)MAXROWS * 4;       // bf16[MAXROWS*INTER]
constexpr size_t WS_UPT   = WS_HBUF + (size_t)MAXROWS * INTER * 2;            // bf16[E][INTER][HIDDEN]
constexpr size_t WS_DWT   = WS_UPT  + (size_t)NUM_EXPERTS * INTER * HIDDEN * 2; // bf16[E][HIDDEN][INTER]

__device__ __forceinline__ unsigned short f2bf(float f) {
    unsigned int u = __float_as_uint(f);
    unsigned int r = u + 0x7FFFu + ((u >> 16) & 1u);   // round-to-nearest-even
    return (unsigned short)(r >> 16);
}

// ---------------- 1: init ----------------
__global__ void moe_init_kernel(float* __restrict__ out, int* cnt, int* cur, float* P) {
    size_t i = (size_t)blockIdx.x * 256 + threadIdx.x;
    if (i < (size_t)NTOK * HIDDEN) out[i] = 0.0f;
    if (i < NUM_EXPERTS) { cnt[i] = 0; cur[i] = 0; P[i] = 0.0f; }
}

// ---------------- 2: weight transpose + fp32->bf16 convert ----------------
// src: fp32 [E][K][N]  ->  dst: bf16 [E][N][K]   (B fragments want per-column contiguous K)
__global__ __launch_bounds__(256)
void moe_transpose_cvt_kernel(const float* __restrict__ src, unsigned short* __restrict__ dst,
                              int K, int N) {
    __shared__ float tile[32][33];
    const int e = blockIdx.z;
    const float* s = src + (size_t)e * K * N;
    unsigned short* d = dst + (size_t)e * K * N;
    const int kt = blockIdx.y * 32, nt = blockIdx.x * 32;
    const int tx = threadIdx.x & 31, ty = threadIdx.x >> 5;   // 32 x 8
    #pragma unroll
    for (int i = 0; i < 32; i += 8)
        tile[ty + i][tx] = s[(size_t)(kt + ty + i) * N + nt + tx];
    __syncthreads();
    #pragma unroll
    for (int i = 0; i < 32; i += 8)
        d[(size_t)(nt + ty + i) * K + kt + tx] = f2bf(tile[tx][ty + i]);
}

// ---------------- 3: router (one wave per token) ----------------
__global__ __launch_bounds__(256)
void moe_router_kernel(const float* __restrict__ x, const float* __restrict__ rw,
                       int* __restrict__ tki, float* __restrict__ tkw,
                       int* __restrict__ cnt, float* __restrict__ Psum) {
    __shared__ float sP[NUM_EXPERTS];
    __shared__ int   sC[NUM_EXPERTS];
    int tid = threadIdx.x;
    if (tid < NUM_EXPERTS) { sP[tid] = 0.0f; sC[tid] = 0; }
    __syncthreads();

    int wave = tid >> 5, lane = tid & 31;
    int tbase = blockIdx.x * 64 + wave * 8;
    for (int it = 0; it < 8; ++it) {
        int t = tbase + it;
        float acc[NUM_EXPERTS];
        #pragma unroll
        for (int e = 0; e < NUM_EXPERTS; ++e) acc[e] = 0.0f;
        for (int h = lane; h < HIDDEN; h += 32) {
            float xv = x[(size_t)t * HIDDEN + h];
            #pragma unroll
            for (int e = 0; e < NUM_EXPERTS; ++e)
                acc[e] = fmaf(xv, rw[e * HIDDEN + h], acc[e]);
        }
        #pragma unroll
        for (int e = 0; e < NUM_EXPERTS; ++e)
            for (int m = 16; m > 0; m >>= 1) acc[e] += __shfl_xor(acc[e], m, 32);

        if (lane == 0) {
            float mx = acc[0];
            #pragma unroll
            for (int e = 1; e < NUM_EXPERTS; ++e) mx = fmaxf(mx, acc[e]);
            float p[NUM_EXPERTS], s = 0.0f;
            #pragma unroll
            for (int e = 0; e < NUM_EXPERTS; ++e) { p[e] = expf(acc[e] - mx); s += p[e]; }
            float inv = 1.0f / s;
            int e0 = 0;
            #pragma unroll
            for (int e = 1; e < NUM_EXPERTS; ++e) if (p[e] > p[e0]) e0 = e;
            int e1 = (e0 == 0) ? 1 : 0;
            #pragma unroll
            for (int e = 0; e < NUM_EXPERTS; ++e)
                if (e != e0 && p[e] > p[e1]) e1 = e;
            float wsum = 1.0f / (p[e0] + p[e1]);           // normalized top-k ratio
            tki[t * 2 + 0] = e0;  tkw[t * 2 + 0] = p[e0] * wsum;
            tki[t * 2 + 1] = e1;  tkw[t * 2 + 1] = p[e1] * wsum;
            atomicAdd(&sC[e0], 1);
            atomicAdd(&sC[e1], 1);
            #pragma unroll
            for (int e = 0; e < NUM_EXPERTS; ++e) atomicAdd(&sP[e], p[e] * inv);
        }
    }
    __syncthreads();
    if (tid < NUM_EXPERTS) { atomicAdd(&cnt[tid], sC[tid]); atomicAdd(&Psum[tid], sP[tid]); }
}

// ---------------- 4: bucket offsets (BM-aligned so tiles never straddle experts) ----------------
__global__ void moe_offsets_kernel(const int* __restrict__ cnt, int* __restrict__ off) {
    if (threadIdx.x == 0 && blockIdx.x == 0) {
        int o = 0;
        for (int e = 0; e < NUM_EXPERTS; ++e) {
            off[e] = o;
            o += ((cnt[e] + BM - 1) / BM) * BM;
        }
        off[NUM_EXPERTS] = o;
    }
}

// ---------------- 5: scatter tokens into expert buckets ----------------
__global__ void moe_scatter_kernel(const int* __restrict__ tki, const float* __restrict__ tkw,
                                   const int* __restrict__ off, int* __restrict__ cur,
                                   int* __restrict__ rtok, float* __restrict__ rgate) {
    int t = blockIdx.x * blockDim.x + threadIdx.x;
    if (t >= NTOK) return;
    #pragma unroll
    for (int k = 0; k < TOPK; ++k) {
        int e = tki[t * 2 + k];
        int pos = atomicAdd(&cur[e], 1);
        int row = off[e] + pos;
        rtok[row]  = t;
        rgate[row] = tkw[t * 2 + k];
    }
}

// ---- WMMA fragment helper (ISA 7.12.2 bf16 layouts: two 16B runs per lane) ----
__device__ __forceinline__ bf16x16 load_frag(const unsigned short* p0, const unsigned short* p1) {
    union { us8 h[2]; bf16x16 v; } u;
    u.h[0] = *(const us8*)p0;
    u.h[1] = *(const us8*)p1;
    return u.v;
}

// ---------------- 6: up-proj GEMM + GELU, bf16 WMMA ----------------
__global__ __launch_bounds__(GEMM_THREADS)
void moe_up_gemm_gelu(const float* __restrict__ x, const unsigned short* __restrict__ upwT,
                      const int* __restrict__ rtok, const int* __restrict__ off,
                      const int* __restrict__ cnt, unsigned short* __restrict__ Hbuf) {
    const int e = blockIdx.z;
    const int cn = cnt[e];
    const int rowTile = blockIdx.y * BM;
    if (rowTile >= cn) return;
    const int base  = off[e];
    const int nTile = blockIdx.x * BN;

    __shared__ unsigned short As[2][BM * LDA];   // 2 x 20.0 KB
    __shared__ unsigned short Bs[2][BN * LDB];   // 2 x 10.0 KB

    const int tid = threadIdx.x, lane = tid & 31, wave = tid >> 5;
    const int wm = wave & 3, wn = wave >> 2;           // 4x4 waves: 64 rows x 32 cols each
    const int lhalf = (lane < 16) ? 0 : 1;

    f32x8 acc[4][2] = {};

    const unsigned short* wbase = upwT + (size_t)e * INTER * HIDDEN + (size_t)nTile * HIDDEN;

    // ---- loader setup, hoisted out of the k-loop (row->token map is k-invariant) ----
    const int ac   = (tid & 7) * 4;              // A col group within BK
    const int arow = tid >> 3;                   // A row within 64-row group
    const float* aptr[4];
    bool aval[4];
    #pragma unroll
    for (int it = 0; it < 4; ++it) {
        int gr = rowTile + it * 64 + arow;
        aval[it] = (gr < cn);
        int tok = aval[it] ? rtok[base + gr] : 0;
        aptr[it] = x + (size_t)tok * HIDDEN + ac;
    }
    const int bn = tid >> 2;                     // B row (column of W)
    const int bc = (tid & 3) * 8;                // 8 bf16 per thread
    const unsigned short* bptr = wbase + (size_t)bn * HIDDEN + bc;

    float4 areg[4];
    us8    breg;

    // ---- prologue: fetch and stage tile 0 ----
    #pragma unroll
    for (int it = 0; it < 4; ++it)
        areg[it] = aval[it] ? *(const float4*)(aptr[it]) : make_float4(0.f, 0.f, 0.f, 0.f);
    breg = *(const us8*)(bptr);
    #pragma unroll
    for (int it = 0; it < 4; ++it) {
        unsigned short* d = &As[0][(it * 64 + arow) * LDA + ac];
        d[0] = f2bf(areg[it].x); d[1] = f2bf(areg[it].y);
        d[2] = f2bf(areg[it].z); d[3] = f2bf(areg[it].w);
    }
    *(us8*)&Bs[0][bn * LDB + bc] = breg;
    __syncthreads();

    constexpr int NK = HIDDEN / BK;
    for (int ki = 0; ki < NK; ++ki) {
        const int cur = ki & 1;
        const bool has_next = (ki + 1) < NK;
        if (has_next) {                          // issue next tile's global loads early
            const int kb = (ki + 1) * BK;
            #pragma unroll
            for (int it = 0; it < 4; ++it)
                areg[it] = aval[it] ? *(const float4*)(aptr[it] + kb)
                                    : make_float4(0.f, 0.f, 0.f, 0.f);
            breg = *(const us8*)(bptr + kb);
        }

        bf16x16 afr[4], bfr[2];
        #pragma unroll
        for (int mf = 0; mf < 4; ++mf) {
            const unsigned short* p = &As[cur][(wm * 64 + mf * 16 + (lane & 15)) * LDA + lhalf * 8];
            afr[mf] = load_frag(p, p + 16);
        }
        #pragma unroll
        for (int nf = 0; nf < 2; ++nf) {
            const unsigned short* p = &Bs[cur][(wn * 32 + nf * 16 + (lane & 15)) * LDB + lhalf * 8];
            bfr[nf] = load_frag(p, p + 16);
        }
        #pragma unroll
        for (int mf = 0; mf < 4; ++mf)
            #pragma unroll
            for (int nf = 0; nf < 2; ++nf)
                acc[mf][nf] = __builtin_amdgcn_wmma_f32_16x16x32_bf16(
                    false, afr[mf], false, bfr[nf], (short)0, acc[mf][nf], false, false);

        if (has_next) {                          // stage next tile into the other buffer
            const int nb = cur ^ 1;
            #pragma unroll
            for (int it = 0; it < 4; ++it) {
                unsigned short* d = &As[nb][(it * 64 + arow) * LDA + ac];
                d[0] = f2bf(areg[it].x); d[1] = f2bf(areg[it].y);
                d[2] = f2bf(areg[it].z); d[3] = f2bf(areg[it].w);
            }
            *(us8*)&Bs[nb][bn * LDB + bc] = breg;
        }
        __syncthreads();
    }

    // epilogue: exact-erf GELU, store bf16 H
    const int colBase = nTile + wn * 32 + (lane & 15);
    const int rowBase = rowTile + wm * 64 + ((lane < 16) ? 0 : 8);
    #pragma unroll
    for (int mf = 0; mf < 4; ++mf)
        #pragma unroll
        for (int nf = 0; nf < 2; ++nf)
            #pragma unroll
            for (int r = 0; r < 8; ++r) {
                float v = acc[mf][nf][r];
                float g = 0.5f * v * (1.0f + erff(v * 0.70710678118654752f));
                int row = rowBase + mf * 16 + r;
                int col = colBase + nf * 16;
                Hbuf[(size_t)(base + row) * INTER + col] = f2bf(g);
            }
}

// ---------------- 7: down-proj GEMM + gated scatter-add ----------------
__global__ __launch_bounds__(GEMM_THREADS)
void moe_down_gemm_scatter(const unsigned short* __restrict__ Hbuf,
                           const unsigned short* __restrict__ dwT,
                           const int* __restrict__ rtok, const float* __restrict__ rgate,
                           const int* __restrict__ off, const int* __restrict__ cnt,
                           float* __restrict__ out) {
    const int e = blockIdx.z;
    const int cn = cnt[e];
    const int rowTile = blockIdx.y * BM;
    if (rowTile >= cn) return;
    const int base  = off[e];
    const int nTile = blockIdx.x * BN;

    __shared__ unsigned short As[2][BM * LDA];
    __shared__ unsigned short Bs[2][BN * LDB];

    const int tid = threadIdx.x, lane = tid & 31, wave = tid >> 5;
    const int wm = wave & 3, wn = wave >> 2;
    const int lhalf = (lane < 16) ? 0 : 1;

    f32x8 acc[4][2] = {};

    const unsigned short* wbase = dwT + (size_t)e * HIDDEN * INTER + (size_t)nTile * INTER;

    // ---- loader setup (Hbuf rows need no gather/guard: padding rows are zeros) ----
    const int ac   = (tid & 7) * 4;
    const int arow = tid >> 3;
    const unsigned short* aptr[4];
    #pragma unroll
    for (int it = 0; it < 4; ++it)
        aptr[it] = Hbuf + (size_t)(base + rowTile + it * 64 + arow) * INTER + ac;
    const int bn = tid >> 2;
    const int bc = (tid & 3) * 8;
    const unsigned short* bptr = wbase + (size_t)bn * INTER + bc;

    us4 areg[4];
    us8 breg;

    // ---- prologue ----
    #pragma unroll
    for (int it = 0; it < 4; ++it) areg[it] = *(const us4*)(aptr[it]);
    breg = *(const us8*)(bptr);
    #pragma unroll
    for (int it = 0; it < 4; ++it) {
        unsigned short* d = &As[0][(it * 64 + arow) * LDA + ac];
        d[0] = areg[it][0]; d[1] = areg[it][1]; d[2] = areg[it][2]; d[3] = areg[it][3];
    }
    *(us8*)&Bs[0][bn * LDB + bc] = breg;
    __syncthreads();

    constexpr int NK = INTER / BK;
    for (int ki = 0; ki < NK; ++ki) {
        const int cur = ki & 1;
        const bool has_next = (ki + 1) < NK;
        if (has_next) {
            const int kb = (ki + 1) * BK;
            #pragma unroll
            for (int it = 0; it < 4; ++it) areg[it] = *(const us4*)(aptr[it] + kb);
            breg = *(const us8*)(bptr + kb);
        }

        bf16x16 afr[4], bfr[2];
        #pragma unroll
        for (int mf = 0; mf < 4; ++mf) {
            const unsigned short* p = &As[cur][(wm * 64 + mf * 16 + (lane & 15)) * LDA + lhalf * 8];
            afr[mf] = load_frag(p, p + 16);
        }
        #pragma unroll
        for (int nf = 0; nf < 2; ++nf) {
            const unsigned short* p = &Bs[cur][(wn * 32 + nf * 16 + (lane & 15)) * LDB + lhalf * 8];
            bfr[nf] = load_frag(p, p + 16);
        }
        #pragma unroll
        for (int mf = 0; mf < 4; ++mf)
            #pragma unroll
            for (int nf = 0; nf < 2; ++nf)
                acc[mf][nf] = __builtin_amdgcn_wmma_f32_16x16x32_bf16(
                    false, afr[mf], false, bfr[nf], (short)0, acc[mf][nf], false, false);

        if (has_next) {
            const int nb = cur ^ 1;
            #pragma unroll
            for (int it = 0; it < 4; ++it) {
                unsigned short* d = &As[nb][(it * 64 + arow) * LDA + ac];
                d[0] = areg[it][0]; d[1] = areg[it][1]; d[2] = areg[it][2]; d[3] = areg[it][3];
            }
            *(us8*)&Bs[nb][bn * LDB + bc] = breg;
        }
        __syncthreads();
    }

    // epilogue: gated atomic accumulation into output (2 contributions per element)
    const int colBase = nTile + wn * 32 + (lane & 15);
    const int rowBase = rowTile + wm * 64 + ((lane < 16) ? 0 : 8);
    #pragma unroll
    for (int mf = 0; mf < 4; ++mf)
        #pragma unroll
        for (int r = 0; r < 8; ++r) {
            int row = rowBase + mf * 16 + r;
            if (row >= cn) continue;
            int   tok = rtok[base + row];
            float g   = rgate[base + row];
            #pragma unroll
            for (int nf = 0; nf < 2; ++nf) {
                int col = colBase + nf * 16;
                atomicAdd(&out[(size_t)tok * HIDDEN + col], g * acc[mf][nf][r]);
            }
        }
}

// ---------------- 8: aux loss ----------------
__global__ void moe_aux_kernel(const int* __restrict__ cnt, const float* __restrict__ P,
                               float* __restrict__ dst) {
    if (threadIdx.x == 0 && blockIdx.x == 0) {
        float a = 0.0f;
        for (int e = 0; e < NUM_EXPERTS; ++e) {
            float f = (float)cnt[e] / (float)(NTOK * TOPK);
            a += f * (P[e] / (float)NTOK);
        }
        *dst = (float)NUM_EXPERTS * a;
    }
}

// ---------------- launch ----------------
extern "C" void kernel_launch(void* const* d_in, const int* in_sizes, int n_in,
                              void* d_out, int out_size, void* d_ws, size_t ws_size,
                              hipStream_t stream) {
    (void)in_sizes; (void)n_in; (void)out_size; (void)ws_size;
    const float* x   = (const float*)d_in[0];   // hidden_states [2,2048,1024]
    const float* rw  = (const float*)d_in[1];   // router_w [8,1024]
    const float* upw = (const float*)d_in[2];   // up_w [8,1024,2048]
    const float* dww = (const float*)d_in[3];   // down_w [8,2048,1024]
    float* out = (float*)d_out;                 // [4096*1024] output + [1] aux

    char* ws = (char*)d_ws;
    int*            cnt   = (int*)(ws + WS_CNT);
    float*          P     = (float*)(ws + WS_P);
    int*            off   = (int*)(ws + WS_OFF);
    int*            cur   = (int*)(ws + WS_CUR);
    int*            tki   = (int*)(ws + WS_TKI);
    float*          tkw   = (float*)(ws + WS_TKW);
    int*            rtok  = (int*)(ws + WS_RTOK);
    float*          rgate = (float*)(ws + WS_RGATE);
    unsigned short* Hbuf  = (unsigned short*)(ws + WS_HBUF);
    unsigned short* upwT  = (unsigned short*)(ws + WS_UPT);
    unsigned short* dwT   = (unsigned short*)(ws + WS_DWT);

    moe_init_kernel<<<((size_t)NTOK * HIDDEN + 255) / 256, 256, 0, stream>>>(out, cnt, cur, P);
    moe_transpose_cvt_kernel<<<dim3(INTER / 32, HIDDEN / 32, NUM_EXPERTS), 256, 0, stream>>>(
        upw, upwT, HIDDEN, INTER);
    moe_transpose_cvt_kernel<<<dim3(HIDDEN / 32, INTER / 32, NUM_EXPERTS), 256, 0, stream>>>(
        dww, dwT, INTER, HIDDEN);
    moe_router_kernel<<<NTOK / 64, 256, 0, stream>>>(x, rw, tki, tkw, cnt, P);
    moe_offsets_kernel<<<1, 32, 0, stream>>>(cnt, off);
    moe_scatter_kernel<<<(NTOK + 255) / 256, 256, 0, stream>>>(tki, tkw, off, cur, rtok, rgate);
    moe_up_gemm_gelu<<<dim3(INTER / BN, NTOK * TOPK / BM, NUM_EXPERTS), GEMM_THREADS, 0, stream>>>(
        x, upwT, rtok, off, cnt, Hbuf);
    moe_down_gemm_scatter<<<dim3(HIDDEN / BN, NTOK * TOPK / BM, NUM_EXPERTS), GEMM_THREADS, 0, stream>>>(
        Hbuf, dwT, rtok, rgate, off, cnt, out);
    moe_aux_kernel<<<1, 32, 0, stream>>>(cnt, P, out + (size_t)NTOK * HIDDEN);
}
